// SPNNet_24309514895913
// MI455X (gfx1250) — matrix operand order repaired and checked by reading, not compiled, not run on hardware
//
#include <hip/hip_runtime.h>
#include <math.h>

// ---------------------------------------------------------------------------
// SPN net for MI455X (gfx1250, wave32, WMMA, async-to-LDS staging).
// k1: h = relu(x @ W1^T + b1) via v_wmma_f32_16x16x32_bf16. Per-wave-private
//     double-buffered global_load_async_to_lds_b128 staging of x (no block
//     barriers in the K loop), fused SPN layer 1 (pre-BN) + per-block partial
//     sums for batch statistics.
// k2: finalize BN1 scale/shift.
// k3: SPN layer 2 (pre-BN) -> d_out (as scratch) + partial sums.
// k4: finalize BN2 scale/shift.
// k5: BN2 affine + 10x10 head + log_softmax, in-place on d_out.
// All reductions fixed-order (deterministic), no atomics.
// ---------------------------------------------------------------------------

typedef __attribute__((ext_vector_type(16))) __bf16 v16bf;
typedef __attribute__((ext_vector_type(8)))  float  v8f;

#define B_ROWS 32768
#define F_IN   784
#define H1     32
#define O1     16
#define P1     16
#define O2     10
#define P2     8
#define NOUT   10
#define NBLK1  512                    // B_ROWS / 64
#define NBLK3  128                    // B_ROWS / 256
#define HALF_LOG2PI 0.91893853320467274f

__device__ __forceinline__ float lse2(float a, float b) {
  float m = fmaxf(a, b);
  return m + __logf(__expf(a - m) + __expf(b - m));
}

__device__ __forceinline__ v16bf pack16(float4 a, float4 b, float4 c, float4 d,
                                        float m) {
  v16bf r;
  r[0]  = (__bf16)(a.x * m); r[1]  = (__bf16)(a.y * m);
  r[2]  = (__bf16)(a.z * m); r[3]  = (__bf16)(a.w * m);
  r[4]  = (__bf16)(b.x * m); r[5]  = (__bf16)(b.y * m);
  r[6]  = (__bf16)(b.z * m); r[7]  = (__bf16)(b.w * m);
  r[8]  = (__bf16)(c.x * m); r[9]  = (__bf16)(c.y * m);
  r[10] = (__bf16)(c.z * m); r[11] = (__bf16)(c.w * m);
  r[12] = (__bf16)(d.x * m); r[13] = (__bf16)(d.y * m);
  r[14] = (__bf16)(d.z * m); r[15] = (__bf16)(d.w * m);
  return r;
}

// issue async staging of one 16x32 f32 sub-tile (this wave's rows) into LDS
__device__ __forceinline__ void issue_tile(const float* __restrict__ x, int rowb,
                                           int kbase, unsigned ldsdst, int lane) {
#pragma unroll
  for (int i = 0; i < 4; ++i) {
    int s = lane + i * 32;             // 0..127 b128 slots (16 rows x 8)
    const float* gp = x + (size_t)(rowb + (s >> 3)) * F_IN + kbase + ((s & 7) << 2);
    unsigned la = ldsdst + (unsigned)s * 16u;
    asm volatile("global_load_async_to_lds_b128 %0, %1, off"
                 :: "v"(la), "v"(gp) : "memory");
  }
  // protect address VGPRs of in-flight async ops against WAR reuse
  asm volatile("s_wait_xcnt 0x0" ::: "memory");
}

// one 16x16x32 K-step: A from this wave's LDS sub-tile, B from W1, 2 WMMAs
__device__ __forceinline__ void mma_tile(const float* __restrict__ wtile,
                                         const float* __restrict__ W1,
                                         int n0, int khalfA, float msk, int kc,
                                         v8f& acc0, v8f& acc1) {
  const float4* ap0 = (const float4*)&wtile[n0 * 32 + khalfA];
  const float4* ap1 = (const float4*)&wtile[n0 * 32 + 8 + khalfA];
  v16bf a = pack16(ap0[0], ap0[1], ap1[0], ap1[1], 1.0f);
  const float4* w0p = (const float4*)(W1 + n0 * F_IN + kc);
  const float4* w1p = (const float4*)(W1 + (n0 + 16) * F_IN + kc);
  v16bf bf0 = pack16(w0p[0], w0p[1], w0p[2], w0p[3], msk);
  v16bf bf1 = pack16(w1p[0], w1p[1], w1p[2], w1p[3], msk);
  acc0 = __builtin_amdgcn_wmma_f32_16x16x32_bf16(false, a, false, bf0,
                                                 (short)0, acc0, false, false);
  acc1 = __builtin_amdgcn_wmma_f32_16x16x32_bf16(false, a, false, bf1,
                                                 (short)0, acc1, false, false);
}

// ---------------------------------------------------------------------------
// Kernel 1: 512 blocks x 128 threads (4 waves). Each wave: 16 rows x 32 cols.
// ---------------------------------------------------------------------------
__global__ __launch_bounds__(128)
void k1_gemm_spn1(const float* __restrict__ x, const float* __restrict__ W1,
                  const float* __restrict__ b1, const float* __restrict__ mean1,
                  const float* __restrict__ std1, const float* __restrict__ sw1,
                  const int* __restrict__ scopes1,
                  float* __restrict__ out1, float* __restrict__ part1) {
  __shared__ float lds_x[2][64 * 32];   // 2 buffers, 4 wave-private 16x32 tiles
  __shared__ float red_s[128];
  __shared__ float red_q[128];

  const int tid  = threadIdx.x;
  const int wave = tid >> 5;
  const int lane = tid & 31;
  const int row0 = blockIdx.x * 64;
  const int rowb = row0 + wave * 16;       // this wave's first row

  const int n0     = lane & 15;            // column / local row index
  const int khalfA = (lane >> 4) << 4;     // A-frag: 0 or 16
  const int koffB  = (lane >> 4) << 4;     // B-frag: 0 or 16

  // LDS byte offsets (generic addr low 32 bits == LDS addr); wave-private
  const unsigned ldsbase = (unsigned)(size_t)(&lds_x[0][0]);
  const unsigned wb0 = ldsbase + (unsigned)(wave * 2048);
  const unsigned wb1 = wb0 + 8192u;
  float* wt0 = &lds_x[0][wave * 16 * 32];
  float* wt1 = &lds_x[1][wave * 16 * 32];

  issue_tile(x, rowb, 0, wb0, lane);       // tile 0 -> buf0

  v8f acc0 = {};
  v8f acc1 = {};

  // steady state: tiles 0..21, branchless 2x-unrolled double buffer
  for (int kt = 0; kt < 22; kt += 2) {
    asm volatile("s_wait_dscnt 0x0" ::: "memory");      // buf1 readers done
    issue_tile(x, rowb, (kt + 1) * 32, wb1, lane);      // tile kt+1 -> buf1
    asm volatile("s_wait_asynccnt 0x4" ::: "memory");   // tile kt landed
    mma_tile(wt0, W1, n0, khalfA, 1.0f, kt * 32 + koffB, acc0, acc1);

    asm volatile("s_wait_dscnt 0x0" ::: "memory");      // buf0 readers done
    issue_tile(x, rowb, (kt + 2) * 32, wb0, lane);      // tile kt+2 -> buf0
    asm volatile("s_wait_asynccnt 0x4" ::: "memory");   // tile kt+1 landed
    mma_tile(wt1, W1, n0, khalfA, 1.0f, (kt + 1) * 32 + koffB, acc0, acc1);
  }

  // tile 22 (buf0): issue tile 23 -> buf1
  asm volatile("s_wait_dscnt 0x0" ::: "memory");
  issue_tile(x, rowb, 23 * 32, wb1, lane);
  asm volatile("s_wait_asynccnt 0x4" ::: "memory");
  mma_tile(wt0, W1, n0, khalfA, 1.0f, 22 * 32 + koffB, acc0, acc1);

  // tile 23 (buf1): issue tail tile 24 -> buf0 (zero cols 16..31, load 0..15)
  asm volatile("s_wait_dscnt 0x0" ::: "memory");
  {
    float4 z; z.x = 0.0f; z.y = 0.0f; z.z = 0.0f; z.w = 0.0f;
#pragma unroll
    for (int i = 0; i < 2; ++i) {
      int s = lane + i * 32;             // 0..63 (16 rows x 4 chunks)
      int r = s >> 2, c = s & 3;
      *(float4*)&wt0[r * 32 + 16 + c * 4] = z;
      const float* gp = x + (size_t)(rowb + r) * F_IN + 768 + (c << 2);
      unsigned la = wb0 + (unsigned)((r * 32 + c * 4) * 4);
      asm volatile("global_load_async_to_lds_b128 %0, %1, off"
                   :: "v"(la), "v"(gp) : "memory");
    }
    asm volatile("s_wait_xcnt 0x0" ::: "memory");
  }
  asm volatile("s_wait_asynccnt 0x2" ::: "memory");     // tile 23 landed
  mma_tile(wt1, W1, n0, khalfA, 1.0f, 23 * 32 + koffB, acc0, acc1);

  // tile 24 (buf0, tail): only lane-half with k=768..783 is valid
  asm volatile("s_wait_asynccnt 0x0" ::: "memory");
  {
    const float mskT = (lane < 16) ? 1.0f : 0.0f;
    const int   kcT  = (lane < 16) ? 768 : 0;           // clamp keeps in-bounds
    mma_tile(wt0, W1, n0, khalfA, mskT, kcT, acc0, acc1);
  }

  // bias + relu, spill h tile to LDS (C/D layout: VGPR r -> M = r + 8*(lane>=16))
  float* htile = &lds_x[0][0];
  const float bias0 = b1[n0], bias1 = b1[n0 + 16];
#pragma unroll
  for (int r = 0; r < 8; ++r) {
    int rl = wave * 16 + r + ((lane >> 4) << 3);
    htile[rl * 32 + n0]      = fmaxf(acc0[r] + bias0, 0.0f);
    htile[rl * 32 + n0 + 16] = fmaxf(acc1[r] + bias1, 0.0f);
  }
  __syncthreads();   // first block barrier: share h across waves

  // SPN layer 1 (pre-BN). Thread t: output o = t&15, rows (t>>4)+8i.
  // Loop order p-outer: per-(o,p) params hoisted out of the 8-row inner loop.
  const int o     = tid & 15;
  const int rbase = tid >> 4;              // 0..7
  float sacc[8];
#pragma unroll
  for (int i = 0; i < 8; ++i) sacc[i] = 0.0f;

  for (int p = 0; p < P1; ++p) {
    int f0 = scopes1[o * H1 + 2 * p];
    int f1 = scopes1[o * H1 + 2 * p + 1];
    int base0 = ((o * P1 + p) * 2 + 0) * 2;
    int base1 = base0 + 2;
    float mu00 = mean1[base0],     mu01 = mean1[base0 + 1];
    float mu10 = mean1[base1],     mu11 = mean1[base1 + 1];
    float i00  = 1.0f / std1[base0],     i01 = 1.0f / std1[base0 + 1];
    float i10  = 1.0f / std1[base1],     i11 = 1.0f / std1[base1 + 1];
    float c0 = __logf(i00) + __logf(i01) - 2.0f * HALF_LOG2PI;
    float c1 = __logf(i10) + __logf(i11) - 2.0f * HALF_LOG2PI;
    float w0 = sw1[(o * P1 + p) * 2], w1 = sw1[(o * P1 + p) * 2 + 1];
    float lz  = lse2(w0, w1);
    float lw0 = w0 - lz + c0;
    float lw1 = w1 - lz + c1;
#pragma unroll
    for (int i = 0; i < 8; ++i) {
      const float* hrow = &htile[(rbase + 8 * i) * 32];
      float x0 = hrow[f0], x1 = hrow[f1];
      float z00 = (x0 - mu00) * i00, z01 = (x1 - mu01) * i01;
      float z10 = (x0 - mu10) * i10, z11 = (x1 - mu11) * i11;
      float p0 = lw0 - 0.5f * (z00 * z00 + z01 * z01);
      float p1 = lw1 - 0.5f * (z10 * z10 + z11 * z11);
      sacc[i] += lse2(p0, p1);
    }
  }

  float sum_l = 0.0f, sq_l = 0.0f;
#pragma unroll
  for (int i = 0; i < 8; ++i) {
    int rl = rbase + 8 * i;
    out1[(size_t)(row0 + rl) * O1 + o] = sacc[i];
    sum_l += sacc[i];
    sq_l  += sacc[i] * sacc[i];
  }
  red_s[tid] = sum_l;
  red_q[tid] = sq_l;
  __syncthreads();
  if (tid < 16) {
    float ts = 0.0f, tq = 0.0f;
    for (int j = 0; j < 8; ++j) { ts += red_s[tid + 16 * j]; tq += red_q[tid + 16 * j]; }
    part1[blockIdx.x * 32 + tid]      = ts;
    part1[blockIdx.x * 32 + 16 + tid] = tq;
  }
}

// ---------------------------------------------------------------------------
// Kernel 2: finalize BN1 -> scale/shift [32]
// ---------------------------------------------------------------------------
__global__ void k2_bn1(const float* __restrict__ part1, const float* __restrict__ bn_w,
                       const float* __restrict__ bn_b, float* __restrict__ ss) {
  int o = threadIdx.x;
  if (o >= O1) return;
  float s = 0.0f, q = 0.0f;
  for (int b = 0; b < NBLK1; ++b) {
    s += part1[b * 32 + o];
    q += part1[b * 32 + 16 + o];
  }
  float mu  = s / (float)B_ROWS;
  float var = q / (float)B_ROWS - mu * mu;
  float sc  = bn_w[o] * rsqrtf(var + 1e-5f);
  ss[o]      = sc;
  ss[O1 + o] = bn_b[o] - mu * sc;
}

// ---------------------------------------------------------------------------
// Kernel 3: SPN layer 2 (pre-BN) -> d_out (scratch), + partial sums.
// ---------------------------------------------------------------------------
__global__ __launch_bounds__(256)
void k3_spn2(const float* __restrict__ out1, const float* __restrict__ bn1ss,
             const float* __restrict__ mean2, const float* __restrict__ std2,
             const float* __restrict__ sw2, const int* __restrict__ scopes2,
             float* __restrict__ out2, float* __restrict__ part2) {
  __shared__ float wred[8 * 20];           // per-wave partials: [wave][sum10,sq10]
  const int tid = threadIdx.x;
  const int wid = tid >> 5;
  const int ln  = tid & 31;
  const int row = blockIdx.x * 256 + tid;

  float s2[O2];
  for (int oo = 0; oo < O2; ++oo) {
    float s = 0.0f;
    for (int p = 0; p < P2; ++p) {
      int f0 = scopes2[oo * O1 + 2 * p];
      int f1 = scopes2[oo * O1 + 2 * p + 1];
      float x0 = out1[(size_t)row * O1 + f0] * bn1ss[f0] + bn1ss[O1 + f0];
      float x1 = out1[(size_t)row * O1 + f1] * bn1ss[f1] + bn1ss[O1 + f1];
      float pr[2];
#pragma unroll
      for (int m = 0; m < 2; ++m) {
        int base = ((oo * P2 + p) * 2 + m) * 2;
        float z0 = (x0 - mean2[base])     / std2[base];
        float z1 = (x1 - mean2[base + 1]) / std2[base + 1];
        pr[m] = -0.5f * (z0 * z0 + z1 * z1)
                - __logf(std2[base]) - __logf(std2[base + 1])
                - 2.0f * HALF_LOG2PI;
      }
      float w0 = sw2[(oo * P2 + p) * 2], w1 = sw2[(oo * P2 + p) * 2 + 1];
      float lz = lse2(w0, w1);
      s += lse2(pr[0] + w0 - lz, pr[1] + w1 - lz);
    }
    s2[oo] = s;
    out2[(size_t)row * O2 + oo] = s;
  }
  // wave-level fixed-order shuffle reductions, one barrier, 8-way final sum
  for (int oo = 0; oo < O2; ++oo) {
    float v = s2[oo];
    float q = s2[oo] * s2[oo];
#pragma unroll
    for (int d = 16; d > 0; d >>= 1) {
      v += __shfl_down(v, d, 32);
      q += __shfl_down(q, d, 32);
    }
    if (ln == 0) {
      wred[wid * 20 + oo]      = v;
      wred[wid * 20 + 10 + oo] = q;
    }
  }
  __syncthreads();
  if (tid < 20) {
    float t = 0.0f;
    for (int w = 0; w < 8; ++w) t += wred[w * 20 + tid];
    part2[blockIdx.x * 20 + tid] = t;
  }
}

// ---------------------------------------------------------------------------
// Kernel 4: finalize BN2 -> scale/shift [20]
// ---------------------------------------------------------------------------
__global__ void k4_bn2(const float* __restrict__ part2, const float* __restrict__ bn_w,
                       const float* __restrict__ bn_b, float* __restrict__ ss) {
  int o = threadIdx.x;
  if (o >= O2) return;
  float s = 0.0f, q = 0.0f;
  for (int b = 0; b < NBLK3; ++b) {
    s += part2[b * 20 + o];
    q += part2[b * 20 + 10 + o];
  }
  float mu  = s / (float)B_ROWS;
  float var = q / (float)B_ROWS - mu * mu;
  float sc  = bn_w[o] * rsqrtf(var + 1e-5f);
  ss[o]      = sc;
  ss[O2 + o] = bn_b[o] - mu * sc;
}

// ---------------------------------------------------------------------------
// Kernel 5: BN2 affine + head GEMV (10x10) + log_softmax, in-place on d_out.
// ---------------------------------------------------------------------------
__global__ __launch_bounds__(256)
void k5_head(float* __restrict__ out, const float* __restrict__ bn2ss,
             const float* __restrict__ W2, const float* __restrict__ b2) {
  const int row = blockIdx.x * 256 + threadIdx.x;
  float y[O2];
#pragma unroll
  for (int o = 0; o < O2; ++o)
    y[o] = out[(size_t)row * O2 + o] * bn2ss[o] + bn2ss[O2 + o];

  float lg[NOUT];
  float mx = -1e30f;
#pragma unroll
  for (int j = 0; j < NOUT; ++j) {
    float s = b2[j];
#pragma unroll
    for (int o = 0; o < O2; ++o) s += y[o] * W2[j * O2 + o];
    lg[j] = s;
    mx = fmaxf(mx, s);
  }
  float se = 0.0f;
#pragma unroll
  for (int j = 0; j < NOUT; ++j) se += __expf(lg[j] - mx);
  float lse = mx + __logf(se);
#pragma unroll
  for (int j = 0; j < NOUT; ++j) out[(size_t)row * O2 + j] = lg[j] - lse;
}

// ---------------------------------------------------------------------------
extern "C" void kernel_launch(void* const* d_in, const int* in_sizes, int n_in,
                              void* d_out, int out_size, void* d_ws, size_t ws_size,
                              hipStream_t stream) {
  const float* x      = (const float*)d_in[0];
  const float* W1     = (const float*)d_in[1];
  const float* b1     = (const float*)d_in[2];
  const float* mean1  = (const float*)d_in[3];
  const float* std1   = (const float*)d_in[4];
  const float* sw1    = (const float*)d_in[5];
  const float* bn1_w  = (const float*)d_in[6];
  const float* bn1_b  = (const float*)d_in[7];
  const float* mean2  = (const float*)d_in[8];
  const float* std2   = (const float*)d_in[9];
  const float* sw2    = (const float*)d_in[10];
  const float* bn2_w  = (const float*)d_in[11];
  const float* bn2_b  = (const float*)d_in[12];
  const float* W2     = (const float*)d_in[13];
  const float* b2     = (const float*)d_in[14];
  const int* scopes1  = (const int*)d_in[15];
  const int* scopes2  = (const int*)d_in[16];

  float* ws    = (float*)d_ws;
  float* out1  = ws;                              // [B, 16]
  float* part1 = out1 + (size_t)B_ROWS * O1;      // [512, 32]
  float* bn1ss = part1 + NBLK1 * 32;              // [32]
  float* part2 = bn1ss + 32;                      // [128, 20]
  float* bn2ss = part2 + NBLK3 * 20;              // [20]
  float* outp  = (float*)d_out;                   // [B, 10]

  k1_gemm_spn1<<<NBLK1, 128, 0, stream>>>(x, W1, b1, mean1, std1, sw1, scopes1,
                                          out1, part1);
  k2_bn1<<<1, 32, 0, stream>>>(part1, bn1_w, bn1_b, bn1ss);
  k3_spn2<<<NBLK3, 256, 0, stream>>>(out1, bn1ss, mean2, std2, sw2, scopes2,
                                     outp, part2);
  k4_bn2<<<1, 32, 0, stream>>>(part2, bn2_w, bn2_b, bn2ss);
  k5_head<<<NBLK3, 256, 0, stream>>>(outp, bn2ss, W2, b2);
}
